// LlamaAttention_17703855194418
// MI455X (gfx1250) — matrix operand order repaired and checked by reading, not compiled, not run on hardware
//
#include <hip/hip_runtime.h>
#include <hip/hip_bf16.h>

#define HIDDEN   4096
#define N_HEADS  32
#define N_KVH    8
#define HEAD_DIM 128
#define SEQ      2048
#define KV_DIM   (N_KVH * HEAD_DIM)   // 1024

typedef __attribute__((ext_vector_type(16))) __bf16 v16bf;
typedef __attribute__((ext_vector_type(8)))  float  v8f;
typedef __attribute__((ext_vector_type(4)))  unsigned int v4u;
typedef __attribute__((ext_vector_type(8)))  int v8i;
typedef __attribute__((ext_vector_type(4)))  int v4i;

union Frag { uint4 u[2]; v16bf v; };

__device__ __forceinline__ v8f wmma_bf16(v16bf a, v16bf b, v8f c) {
  return __builtin_amdgcn_wmma_f32_16x16x32_bf16(false, a, false, b,
                                                 (short)0, c, false, false);
}

__device__ __forceinline__ unsigned short f2bf(float x) {
  unsigned int u = __float_as_uint(x);
  u += 0x7FFFu + ((u >> 16) & 1u);          // round-to-nearest-even
  return (unsigned short)(u >> 16);
}
__device__ __forceinline__ float bf2f(unsigned short h) {
  return __uint_as_float(((unsigned int)h) << 16);
}

// ---- Tensor Data Mover: 2D tile (bf16) global -> LDS ----------------------
// D# packing per CDNA5 ISA (group0 128b, group1 256b).
// 6-arg builtin form: (g0, g1, g2, g3, g4, cpol).
__device__ __forceinline__ void tdm_load_2d(unsigned lds_addr, const void* gptr,
                                            unsigned tensor_dim0, unsigned tensor_dim1,
                                            unsigned tile0, unsigned tile1,
                                            unsigned long long stride0) {
  unsigned long long ga = (unsigned long long)gptr;
  v4u g0;
  g0.x = 1u;                                        // count=1 (user mode)
  g0.y = lds_addr;                                  // lds_addr [63:32]
  g0.z = (unsigned)(ga & 0xFFFFFFFFu);              // global_addr [95:64]
  g0.w = (unsigned)((ga >> 32) & 0x1FFFFFFu)        // global_addr [120:96]
       | (2u << 30);                                // type=2 ("image")
  v8i g1;
  g1[0] = (int)(1u << 16);                          // data_size=1 -> 2 bytes
  g1[1] = (int)((tensor_dim0 & 0xFFFFu) << 16);     // tensor_dim0 [79:48]
  g1[2] = (int)((tensor_dim0 >> 16) | ((tensor_dim1 & 0xFFFFu) << 16));
  g1[3] = (int)((tensor_dim1 >> 16) | (tile0 << 16));  // tile_dim0 [127:112]
  g1[4] = (int)(tile1 & 0xFFFFu);                   // tile_dim1 [143:128]
  g1[5] = (int)(stride0 & 0xFFFFFFFFu);             // tensor_dim0_stride [207:160]
  g1[6] = (int)((stride0 >> 32) & 0xFFFFu);
  g1[7] = 0;
  v4i z4 = {0, 0, 0, 0};
  v8i z8 = {0, 0, 0, 0, 0, 0, 0, 0};
  __builtin_amdgcn_tensor_load_to_lds(g0, g1, z4, z4, z8, 0);
}

// ---------------------------------------------------------------- convert
__global__ void to_bf16_kernel(const float* __restrict__ in,
                               unsigned short* __restrict__ out, int n) {
  int i = blockIdx.x * blockDim.x + threadIdx.x;
  if (i < n) out[i] = f2bf(in[i]);
}

// in[R][C] f32  ->  out[C][R] bf16  (LDS-tiled transpose, block=256)
__global__ void transpose_to_bf16T_kernel(const float* __restrict__ in,
                                          unsigned short* __restrict__ out,
                                          int R, int C) {
  __shared__ float tile[32][33];
  const int c0 = blockIdx.x * 32;
  const int r0 = blockIdx.y * 32;
  const int tx = threadIdx.x & 31;
  const int ty = threadIdx.x >> 5;          // 0..7
  for (int i = ty; i < 32; i += 8)
    tile[i][tx] = in[(size_t)(r0 + i) * C + c0 + tx];
  __syncthreads();
  for (int i = ty; i < 32; i += 8)
    out[(size_t)(c0 + i) * R + r0 + tx] = f2bf(tile[tx][i]);
}

// ---------------------------------------------------------------- GEMM
// C[M][N] = A[M][K] * B[K][N],  A bf16 row-major, B given as Bt[N][K] bf16.
// MODE 0: C bf16 row-major.  MODE 1: C bf16 stored transposed Ct[N][M].
// MODE 2: C f32 row-major.
// block = 256 (8 waves). wave tile = 16x64.  block tile = 32x256.
template <int MODE>
__global__ void __launch_bounds__(256, 1)
gemm_bf16_kernel(const unsigned short* __restrict__ A,
                 const unsigned short* __restrict__ Bt,
                 void* __restrict__ C,
                 int M, int N, int K) {
  const int wave = threadIdx.x >> 5;
  const int lane = threadIdx.x & 31;
  const int l15  = lane & 15;
  const int hi   = lane >> 4;
  const int m0 = blockIdx.y * 32 + (wave & 1) * 16;
  const int n0 = blockIdx.x * 256 + (wave >> 1) * 64;

  v8f acc[4] = {};
  const unsigned short* arow = A + (size_t)(m0 + l15) * K;
  const unsigned short* bcol0 = Bt + (size_t)(n0 + l15) * K;

  for (int kk = 0; kk < K; kk += 32) {
    Frag a, b0, b1, b2, b3;
    const int cb = kk + hi * 16;            // A: lanes0-15 K={0..15}
    const int kb = kk + hi * 16;            // B: lane = column, contiguous k
    a.u[0]  = *(const uint4*)(arow + cb);
    a.u[1]  = *(const uint4*)(arow + cb + 8);
    b0.u[0] = *(const uint4*)(bcol0 + kb);
    b0.u[1] = *(const uint4*)(bcol0 + kb + 8);
    b1.u[0] = *(const uint4*)(bcol0 + (size_t)16 * K + kb);
    b1.u[1] = *(const uint4*)(bcol0 + (size_t)16 * K + kb + 8);
    b2.u[0] = *(const uint4*)(bcol0 + (size_t)32 * K + kb);
    b2.u[1] = *(const uint4*)(bcol0 + (size_t)32 * K + kb + 8);
    b3.u[0] = *(const uint4*)(bcol0 + (size_t)48 * K + kb);
    b3.u[1] = *(const uint4*)(bcol0 + (size_t)48 * K + kb + 8);
    acc[0] = wmma_bf16(a.v, b0.v, acc[0]);
    acc[1] = wmma_bf16(a.v, b1.v, acc[1]);
    acc[2] = wmma_bf16(a.v, b2.v, acc[2]);
    acc[3] = wmma_bf16(a.v, b3.v, acc[3]);
  }

#pragma unroll
  for (int j = 0; j < 4; ++j) {
#pragma unroll
    for (int r = 0; r < 8; ++r) {
      const int row = m0 + r + hi * 8;
      const int col = n0 + j * 16 + l15;
      const float v = acc[j][r];
      if (MODE == 0)      ((unsigned short*)C)[(size_t)row * N + col] = f2bf(v);
      else if (MODE == 1) ((unsigned short*)C)[(size_t)col * M + row] = f2bf(v);
      else                ((float*)C)[(size_t)row * N + col] = v;
    }
  }
}

// ---------------------------------------------------------------- RoPE
// X[s][h*128 + d] in-place. one thread per (s, h, d<64).
__global__ void rope_kernel(unsigned short* __restrict__ X,
                            const int* __restrict__ pos_ids,
                            int nheads) {
  int idx = blockIdx.x * blockDim.x + threadIdx.x;
  int d = idx & 63;
  int t = idx >> 6;
  int h = t % nheads;
  int s = t / nheads;
  if (s >= SEQ) return;
  float p   = (float)pos_ids[s];
  float inv = __expf(-((float)(2 * d) / 128.0f) * 9.210340371976184f);
  float ang = p * inv;
  float c = __cosf(ang), sn = __sinf(ang);
  unsigned short* row = X + (size_t)s * (nheads * HEAD_DIM) + h * HEAD_DIM;
  float x1 = bf2f(row[d]);
  float x2 = bf2f(row[d + 64]);
  row[d]      = f2bf(x1 * c - x2 * sn);
  row[d + 64] = f2bf(x2 * c + x1 * sn);
}

// ---------------------------------------------------------------- attention
// Q[S][4096] bf16, Kc[S][1024] bf16, Vt[1024][S] bf16 -> O[S][4096] bf16.
// grid(S/64, N_HEADS), block 128 (4 waves); wave = one 16-query tile.
// K/V tiles staged in LDS by the Tensor Data Mover (shared by all 4 waves).
__global__ void __launch_bounds__(128, 1)
attn_kernel(const unsigned short* __restrict__ Q,
            const unsigned short* __restrict__ Kc,
            const unsigned short* __restrict__ Vt,
            unsigned short* __restrict__ O) {
  const int h    = blockIdx.y;
  const int kvh  = h >> 2;
  const int wave = threadIdx.x >> 5;
  const int lane = threadIdx.x & 31;
  const int l15  = lane & 15;
  const int hi   = lane >> 4;
  const int q0   = blockIdx.x * 64 + wave * 16;

  __shared__ __align__(16) unsigned short s_k[32][HEAD_DIM];  // [key][d]   8KB
  __shared__ __align__(16) unsigned short s_v[HEAD_DIM][32];  // [d][key]   8KB
  __shared__ __align__(16) float          s_s[4][16][32];     //            8KB
  __shared__ __align__(16) unsigned short s_p[4][16][32];     //            4KB
  __shared__ float s_cf[4][16];
  __shared__ float s_l[4][16];

  // Q fragments for d-chunks of 32 (A-matrix layout)
  Frag qa[4];
  const unsigned short* qrow = Q + (size_t)(q0 + l15) * HIDDEN + h * HEAD_DIM;
#pragma unroll
  for (int c = 0; c < 4; ++c) {
    const int cb = c * 32 + hi * 16;
    qa[c].u[0] = *(const uint4*)(qrow + cb);
    qa[c].u[1] = *(const uint4*)(qrow + cb + 8);
  }

  v8f of[8] = {};
  float m_i = -1e30f, l_i = 0.0f;           // valid on lanes 0..15 (row = lane)
  const float scale = 0.08838834764831845f; // 1/sqrt(128)
  const int ktiles = blockIdx.x * 2 + 2;    // keys covered: blockIdx.x*64 + 64

  for (int kt = 0; kt < ktiles; ++kt) {
    const int k0 = kt * 32;

    // ---- stage K tile [32 keys][128 d] and V tile [128 d][32 keys] via TDM
    __syncthreads();                        // previous-iter LDS reads complete
    if (wave == 0) {
      __builtin_prefetch(Kc + (size_t)(k0 + 32) * KV_DIM + kvh * HEAD_DIM, 0, 1);
      tdm_load_2d((unsigned)(size_t)(void*)s_k,
                  Kc + (size_t)k0 * KV_DIM + kvh * HEAD_DIM,
                  KV_DIM, SEQ, HEAD_DIM, 32, KV_DIM);
      tdm_load_2d((unsigned)(size_t)(void*)s_v,
                  Vt + (size_t)(kvh * HEAD_DIM) * SEQ + k0,
                  SEQ, KV_DIM, 32, HEAD_DIM, SEQ);
      __builtin_amdgcn_s_wait_tensorcnt((short)0);
    }
    __syncthreads();                        // tiles visible to all waves

    // ---- scores: S[16q x 32k] via two 16-key WMMAs x 4 d-chunks (from LDS)
    v8f sc[2] = {};
#pragma unroll
    for (int half = 0; half < 2; ++half) {
      const int kl = half * 16 + l15;       // lane = key column
      const unsigned short* kp = &s_k[kl][hi * 16];
      Frag b0, b1, b2, b3;
      b0.u[0] = *(const uint4*)(kp +  0); b0.u[1] = *(const uint4*)(kp +   8);
      b1.u[0] = *(const uint4*)(kp + 32); b1.u[1] = *(const uint4*)(kp +  40);
      b2.u[0] = *(const uint4*)(kp + 64); b2.u[1] = *(const uint4*)(kp +  72);
      b3.u[0] = *(const uint4*)(kp + 96); b3.u[1] = *(const uint4*)(kp + 104);
      sc[half] = wmma_bf16(qa[0].v, b0.v, sc[half]);
      sc[half] = wmma_bf16(qa[1].v, b1.v, sc[half]);
      sc[half] = wmma_bf16(qa[2].v, b2.v, sc[half]);
      sc[half] = wmma_bf16(qa[3].v, b3.v, sc[half]);
    }

    // ---- dump scores (scaled + causal mask) into LDS
#pragma unroll
    for (int half = 0; half < 2; ++half) {
#pragma unroll
      for (int r = 0; r < 8; ++r) {
        const int row = r + hi * 8;
        const int col = half * 16 + l15;
        const int key = k0 + col;
        s_s[wave][row][col] =
            (key <= q0 + row) ? sc[half][r] * scale : -1e30f;
      }
    }
    __syncthreads();

    // ---- online softmax (lanes 0..15 each own one row)
    if (lane < 16) {
      const int row = lane;
      float mx = m_i;
#pragma unroll
      for (int c = 0; c < 32; ++c) mx = fmaxf(mx, s_s[wave][row][c]);
      const float corr = __expf(m_i - mx);
      float sum = 0.0f;
#pragma unroll
      for (int c = 0; c < 32; ++c) {
        float p = __expf(s_s[wave][row][c] - mx);
        sum += p;
        s_p[wave][row][c] = f2bf(p);
      }
      l_i = l_i * corr + sum;
      m_i = mx;
      s_cf[wave][row] = corr;
    }
    __syncthreads();

    // ---- rescale output accumulators
    float cf8[8];
#pragma unroll
    for (int r = 0; r < 8; ++r) cf8[r] = s_cf[wave][r + hi * 8];
#pragma unroll
    for (int j = 0; j < 8; ++j)
#pragma unroll
      for (int r = 0; r < 8; ++r) of[j][r] *= cf8[r];

    // ---- P fragment (A layout) straight from LDS
    Frag pf;
    pf.u[0] = *(const uint4*)&s_p[wave][l15][hi * 16];
    pf.u[1] = *(const uint4*)&s_p[wave][l15][hi * 16 + 8];

    // ---- O += P[16x32] * V[32x128]  (8 N-chunks of 16, from LDS, groups of 4)
#pragma unroll
    for (int g = 0; g < 2; ++g) {
      Frag v0, v1, v2, v3;
      const unsigned short* vp0 = &s_v[(g * 4 + 0) * 16 + l15][hi * 16];
      const unsigned short* vp1 = &s_v[(g * 4 + 1) * 16 + l15][hi * 16];
      const unsigned short* vp2 = &s_v[(g * 4 + 2) * 16 + l15][hi * 16];
      const unsigned short* vp3 = &s_v[(g * 4 + 3) * 16 + l15][hi * 16];
      v0.u[0] = *(const uint4*)(vp0); v0.u[1] = *(const uint4*)(vp0 + 8);
      v1.u[0] = *(const uint4*)(vp1); v1.u[1] = *(const uint4*)(vp1 + 8);
      v2.u[0] = *(const uint4*)(vp2); v2.u[1] = *(const uint4*)(vp2 + 8);
      v3.u[0] = *(const uint4*)(vp3); v3.u[1] = *(const uint4*)(vp3 + 8);
      of[g * 4 + 0] = wmma_bf16(pf.v, v0.v, of[g * 4 + 0]);
      of[g * 4 + 1] = wmma_bf16(pf.v, v1.v, of[g * 4 + 1]);
      of[g * 4 + 2] = wmma_bf16(pf.v, v2.v, of[g * 4 + 2]);
      of[g * 4 + 3] = wmma_bf16(pf.v, v3.v, of[g * 4 + 3]);
    }
  }

  // ---- finalize: divide by row sums, store bf16
  if (lane < 16) s_l[wave][lane] = l_i;
  __syncthreads();
  float li8[8];
#pragma unroll
  for (int r = 0; r < 8; ++r) li8[r] = 1.0f / s_l[wave][r + hi * 8];
#pragma unroll
  for (int j = 0; j < 8; ++j) {
#pragma unroll
    for (int r = 0; r < 8; ++r) {
      const int row = q0 + r + hi * 8;
      const int col = h * HEAD_DIM + j * 16 + l15;
      O[(size_t)row * HIDDEN + col] = f2bf(of[j][r] * li8[r]);
    }
  }
}

// ---------------------------------------------------------------- launch
extern "C" void kernel_launch(void* const* d_in, const int* in_sizes, int n_in,
                              void* d_out, int out_size, void* d_ws, size_t ws_size,
                              hipStream_t stream) {
  const float* hidden = (const float*)d_in[0];
  const int*   pos    = (const int*)d_in[1];
  const float* Wq     = (const float*)d_in[2];
  const float* Wk     = (const float*)d_in[3];
  const float* Wv     = (const float*)d_in[4];
  const float* Wo     = (const float*)d_in[5];

  char* ws = (char*)d_ws;
  size_t off = 0;
  auto carve = [&](size_t bytes) { char* p = ws + off; off += bytes; return p; };
  unsigned short* Xb  = (unsigned short*)carve((size_t)SEQ * HIDDEN * 2);      // 16 MB
  unsigned short* Wqt = (unsigned short*)carve((size_t)HIDDEN * HIDDEN * 2);   // 32 MB
  unsigned short* Wkt = (unsigned short*)carve((size_t)KV_DIM * HIDDEN * 2);   //  8 MB
  unsigned short* Wvt = (unsigned short*)carve((size_t)KV_DIM * HIDDEN * 2);   //  8 MB
  unsigned short* Wot = (unsigned short*)carve((size_t)HIDDEN * HIDDEN * 2);   // 32 MB
  unsigned short* Qb  = (unsigned short*)carve((size_t)SEQ * HIDDEN * 2);      // 16 MB
  unsigned short* Kb  = (unsigned short*)carve((size_t)SEQ * KV_DIM * 2);      //  4 MB
  unsigned short* Vtb = (unsigned short*)carve((size_t)KV_DIM * SEQ * 2);      //  4 MB
  unsigned short* Ab  = (unsigned short*)carve((size_t)SEQ * HIDDEN * 2);      // 16 MB

  // 1) hidden -> bf16
  {
    int n = SEQ * HIDDEN;
    to_bf16_kernel<<<(n + 255) / 256, 256, 0, stream>>>(hidden, Xb, n);
  }
  // 2) weights -> bf16, transposed (W[K][N] -> Wt[N][K])
  transpose_to_bf16T_kernel<<<dim3(HIDDEN / 32, HIDDEN / 32), 256, 0, stream>>>(
      Wq, Wqt, HIDDEN, HIDDEN);
  transpose_to_bf16T_kernel<<<dim3(KV_DIM / 32, HIDDEN / 32), 256, 0, stream>>>(
      Wk, Wkt, HIDDEN, KV_DIM);
  transpose_to_bf16T_kernel<<<dim3(KV_DIM / 32, HIDDEN / 32), 256, 0, stream>>>(
      Wv, Wvt, HIDDEN, KV_DIM);
  transpose_to_bf16T_kernel<<<dim3(HIDDEN / 32, HIDDEN / 32), 256, 0, stream>>>(
      Wo, Wot, HIDDEN, HIDDEN);

  // 3) projections
  gemm_bf16_kernel<0><<<dim3(HIDDEN / 256, SEQ / 32), 256, 0, stream>>>(
      Xb, Wqt, Qb, SEQ, HIDDEN, HIDDEN);                       // Q = X Wq
  gemm_bf16_kernel<0><<<dim3(KV_DIM / 256, SEQ / 32), 256, 0, stream>>>(
      Xb, Wkt, Kb, SEQ, KV_DIM, HIDDEN);                       // K = X Wk
  gemm_bf16_kernel<1><<<dim3(KV_DIM / 256, SEQ / 32), 256, 0, stream>>>(
      Xb, Wvt, Vtb, SEQ, KV_DIM, HIDDEN);                      // Vt = (X Wv)^T

  // 4) RoPE in place on Q and K
  {
    int nq = SEQ * N_HEADS * 64;
    rope_kernel<<<(nq + 255) / 256, 256, 0, stream>>>(Qb, pos, N_HEADS);
    int nk = SEQ * N_KVH * 64;
    rope_kernel<<<(nk + 255) / 256, 256, 0, stream>>>(Kb, pos, N_KVH);
  }

  // 5) causal flash attention (TDM-staged K/V tiles)
  attn_kernel<<<dim3(SEQ / 64, N_HEADS), 128, 0, stream>>>(Qb, Kb, Vtb, Ab);

  // 6) output projection -> f32 d_out
  gemm_bf16_kernel<2><<<dim3(HIDDEN / 256, SEQ / 32), 256, 0, stream>>>(
      Ab, Wot, d_out, SEQ, HIDDEN, HIDDEN);
}
